// BitLinear_12910671692157
// MI455X (gfx1250) — matrix operand order repaired and checked by reading, not compiled, not run on hardware
//
#include <hip/hip_runtime.h>

// ---------------------------------------------------------------------------
// BitLinear forward for MI455X (gfx1250, wave32, WMMA)
//   y[b,s,o] = sum_i x[b,s,i] * ternary(w)[o,i] + bias[o]
// M = 8192 (B*S), N = 4096 (OUT_F), K = 4096 (IN_F)
// ---------------------------------------------------------------------------

typedef _Float16 v4h  __attribute__((ext_vector_type(4)));
typedef _Float16 v8h  __attribute__((ext_vector_type(8)));
typedef _Float16 v16h __attribute__((ext_vector_type(16)));
typedef float    v4f  __attribute__((ext_vector_type(4)));
typedef float    v8f  __attribute__((ext_vector_type(8)));

#define IN_F   4096
#define OUT_F  4096
#define MROWS  8192   // B * S = 4 * 2048

// ---------------------------------------------------------------------------
// Kernel 1: per-row ternary quantization.
// One 256-thread block per output row. Exact lower-median of |w| found by a
// 31-step binary search over the (monotone) abs-float bit pattern.
// ---------------------------------------------------------------------------
__global__ __launch_bounds__(256) void ternary_quant_kernel(
    const float* __restrict__ w, _Float16* __restrict__ wq)
{
  const int row = blockIdx.x;
  const int t   = threadIdx.x;
  const float* wr = w + (size_t)row * IN_F;

  float    v[16];
  unsigned u[16];
#pragma unroll
  for (int i = 0; i < 16; ++i) {
    v[i] = wr[i * 256 + t];
    u[i] = __float_as_uint(fabsf(v[i]));
  }

  __shared__ int s_cnt;
  unsigned prefix = 0u;
  const int kSel = (IN_F - 1) / 2;  // 2047: torch-style lower median index

  for (int bit = 30; bit >= 0; --bit) {
    const unsigned cand = prefix | (1u << bit);
    if (t == 0) s_cnt = 0;
    __syncthreads();
    int local = 0;
#pragma unroll
    for (int i = 0; i < 16; ++i) local += (u[i] < cand) ? 1 : 0;
    atomicAdd(&s_cnt, local);
    __syncthreads();
    if (s_cnt <= kSel) prefix = cand;  // largest v with count(|w| < v) <= kSel
    __syncthreads();
  }

  const float med   = __uint_as_float(prefix);
  const float s     = fmaxf(med, 1e-12f);
  const float inv_s = 1.0f / s;

  _Float16* o = wq + (size_t)row * IN_F;
#pragma unroll
  for (int i = 0; i < 16; ++i) {
    float q = rintf(v[i] * inv_s);           // round-half-even, like jnp.round
    q = fminf(fmaxf(q, -1.0f), 1.0f);        // ternary {-1,0,1}
    o[i * 256 + t] = (_Float16)(q * s);
  }
}

// ---------------------------------------------------------------------------
// Kernel 2: x fp32 -> f16 (vectorized, streaming).
// ---------------------------------------------------------------------------
__global__ __launch_bounds__(256) void convert_x_kernel(
    const float* __restrict__ x, _Float16* __restrict__ xh, long n4)
{
  long i      = (long)blockIdx.x * blockDim.x + threadIdx.x;
  long stride = (long)gridDim.x * blockDim.x;
  for (; i < n4; i += stride) {
    v4f f = __builtin_nontemporal_load(((const v4f*)x) + i);
    v4h h = { (_Float16)f.x, (_Float16)f.y, (_Float16)f.z, (_Float16)f.w };
    ((v4h*)xh)[i] = h;
  }
}

// ---------------------------------------------------------------------------
// Kernel 3: WMMA GEMM.  C[m,n] = sum_k A[m,k]*B[n,k] + bias[n]
//   A = x_f16   [MROWS][IN_F]  (row-major, K contiguous)
//   B = wq_f16  [OUT_F][IN_F]  (row-major, K contiguous)  -> NT layout
// Block: 256 threads (8 waves), 128x128 C-tile, BK=32 K-slice,
// double-buffered LDS with padded stride (40 halves) for conflict-free b128.
// Wave grid 4(M) x 2(N): each wave owns 32x64 = 2x4 WMMA 16x16 tiles.
// ---------------------------------------------------------------------------
#define BK    32
#define LDST  40            // BK + 8 halves of padding (80B row stride)

__device__ __forceinline__ void tile_gload(const _Float16* __restrict__ base,
                                           int t, int kt, v8h* dst)
{
  // 128 rows x 32 halves = 512 chunks of 8 halves; 2 chunks per thread
#pragma unroll
  for (int j = 0; j < 2; ++j) {
    const int chunk = t + j * 256;
    const int r  = chunk >> 2;          // 0..127
    const int cc = (chunk & 3) * 8;     // 0,8,16,24 halves
    dst[j] = *(const v8h*)(base + (size_t)r * IN_F + (size_t)kt * BK + cc);
  }
}

__device__ __forceinline__ void tile_sstore(_Float16* __restrict__ s,
                                            int t, const v8h* src)
{
#pragma unroll
  for (int j = 0; j < 2; ++j) {
    const int chunk = t + j * 256;
    const int r  = chunk >> 2;
    const int cc = (chunk & 3) * 8;
    *(v8h*)(s + r * LDST + cc) = src[j];
  }
}

__global__ __launch_bounds__(256) void bitlinear_gemm_kernel(
    const _Float16* __restrict__ A,    // [MROWS][IN_F]
    const _Float16* __restrict__ Bq,   // [OUT_F][IN_F]
    const float*    __restrict__ bias, // [OUT_F]
    float*          __restrict__ out)  // [MROWS][OUT_F]
{
  __shared__ _Float16 sA[2][128 * LDST];   // 2 x 10 KB
  __shared__ _Float16 sB[2][128 * LDST];   // 2 x 10 KB   (40 KB total)

  const int t    = threadIdx.x;
  const int lane = t & 31;
  const int wave = t >> 5;       // 0..7
  const int wm   = wave >> 1;    // 0..3 : M direction
  const int wn   = wave & 1;     // 0..1 : N direction
  const int half = lane >> 4;    // 0/1 : lane group (K split per ISA layout)
  const int lr   = lane & 15;

  const size_t m0 = (size_t)blockIdx.y * 128;
  const size_t n0 = (size_t)blockIdx.x * 128;

  const _Float16* Abase = A  + m0 * IN_F;
  const _Float16* Bbase = Bq + n0 * IN_F;

  v8f acc[2][4];
#pragma unroll
  for (int i = 0; i < 2; ++i)
#pragma unroll
    for (int j = 0; j < 4; ++j)
#pragma unroll
      for (int e = 0; e < 8; ++e) acc[i][j][e] = 0.0f;

  v8h ra[2], rb[2];
  tile_gload(Abase, t, 0, ra);
  tile_gload(Bbase, t, 0, rb);
  tile_sstore(sA[0], t, ra);
  tile_sstore(sB[0], t, rb);
  __syncthreads();

  constexpr int NK = IN_F / BK;  // 128 K-slices
  for (int kt = 0; kt < NK; ++kt) {
    const int cur = kt & 1;
    if (kt + 1 < NK) {                      // prefetch next slice into regs
      tile_gload(Abase, t, kt + 1, ra);
      tile_gload(Bbase, t, kt + 1, rb);
    }

    const _Float16* a_s = sA[cur];
    const _Float16* b_s = sB[cur];

    // A fragments: 16-bit A 16x32 layout — lane<16: K 0..7 & 16..23,
    // lane>=16: K 8..15 & 24..31 of row (lane&15)
    v16h af[2];
#pragma unroll
    for (int i = 0; i < 2; ++i) {
      const _Float16* p = a_s + (wm * 32 + i * 16 + lr) * LDST + half * 8;
      v8h lo = *(const v8h*)(p);
      v8h hi = *(const v8h*)(p + 16);
#pragma unroll
      for (int e = 0; e < 8; ++e) { af[i][e] = lo[e]; af[i][e + 8] = hi[e]; }
    }
    // B fragments: 16-bit B 32x16 layout — lane<16: K 0..7 & 16..23 of col (lane&15),
    // lane>=16: K 8..15 & 24..31
    v16h bf[4];
#pragma unroll
    for (int j = 0; j < 4; ++j) {
      const _Float16* p = b_s + (wn * 64 + j * 16 + lr) * LDST + half * 8;
      v8h lo = *(const v8h*)(p);
      v8h hi = *(const v8h*)(p + 16);
#pragma unroll
      for (int e = 0; e < 8; ++e) { bf[j][e] = lo[e]; bf[j][e + 8] = hi[e]; }
    }

#pragma unroll
    for (int i = 0; i < 2; ++i)
#pragma unroll
      for (int j = 0; j < 4; ++j)
        acc[i][j] = __builtin_amdgcn_wmma_f32_16x16x32_f16(
            /*neg_a=*/false, af[i], /*neg_b=*/false, bf[j],
            /*c_mod=*/(short)0, acc[i][j],
            /*reuse_a=*/false, /*reuse_b=*/false);

    if (kt + 1 < NK) {
      tile_sstore(sA[cur ^ 1], t, ra);      // other buffer: no read race
      tile_sstore(sB[cur ^ 1], t, rb);
      __syncthreads();
    }
  }

  // Epilogue: C/D layout — VGPR r: lane<16 -> M=r, lane>=16 -> M=r+8; N=lane&15
#pragma unroll
  for (int i = 0; i < 2; ++i) {
    const size_t mrow0 = m0 + wm * 32 + i * 16 + half * 8;
#pragma unroll
    for (int j = 0; j < 4; ++j) {
      const int col = (int)n0 + wn * 64 + j * 16 + lr;
      const float bv = bias[col];
#pragma unroll
      for (int r = 0; r < 8; ++r) {
        __builtin_nontemporal_store(acc[i][j][r] + bv,
                                    out + (mrow0 + r) * (size_t)OUT_F + col);
      }
    }
  }
}

// ---------------------------------------------------------------------------
// Launch
// ---------------------------------------------------------------------------
extern "C" void kernel_launch(void* const* d_in, const int* in_sizes, int n_in,
                              void* d_out, int out_size, void* d_ws, size_t ws_size,
                              hipStream_t stream) {
  (void)in_sizes; (void)n_in; (void)out_size; (void)ws_size;

  const float* x      = (const float*)d_in[0];   // [4,2048,4096] fp32
  const float* weight = (const float*)d_in[1];   // [4096,4096]   fp32
  const float* bias   = (const float*)d_in[2];   // [4096]        fp32
  float*       out    = (float*)d_out;           // [4,2048,4096] fp32

  // workspace layout: wq_f16 (32 MB) | x_f16 (64 MB)
  _Float16* wq = (_Float16*)d_ws;
  _Float16* xh = (_Float16*)((char*)d_ws + (size_t)OUT_F * IN_F * sizeof(_Float16));

  ternary_quant_kernel<<<OUT_F, 256, 0, stream>>>(weight, wq);

  const long n4 = (long)MROWS * IN_F / 4;
  convert_x_kernel<<<8192, 256, 0, stream>>>(x, xh, n4);

  dim3 grid(OUT_F / 128, MROWS / 128);  // (32, 64)
  bitlinear_gemm_kernel<<<grid, 256, 0, stream>>>(xh, wq, bias, out);
}
